// MLASelfAttention_90108413870572
// MI455X (gfx1250) — matrix operand order repaired and checked by reading, not compiled, not run on hardware
//
#include <hip/hip_runtime.h>
#include <hip/hip_bf16.h>

// ---------------------------------------------------------------------------
// MLA self-attention for MI455X (gfx1250): bf16 WMMA + TDM async tile loads.
//   B=2, T=2048, C=2048, H=16, HS=128, D_LATENT=1024
// Pipeline: cast f32->bf16, 4 GEMMs (q, c_latent, k, v), flash attention,
//           final GEMM (+bias) to f32 output.
// GEMM / attention use double-buffered LDS tiles; the contiguous tiles
// (GEMM A-tile, attention K-tile) are staged by the Tensor Data Mover
// (tensor_load_to_lds, TENSORcnt), transposed tiles are register-staged.
// ---------------------------------------------------------------------------

typedef __bf16 bf16_t;
typedef bf16_t bf16x4 __attribute__((ext_vector_type(4)));
typedef bf16_t bf16x8 __attribute__((ext_vector_type(8)));
typedef bf16_t v16bf  __attribute__((ext_vector_type(16)));
typedef float  v8f    __attribute__((ext_vector_type(8)));

typedef unsigned int u32x4 __attribute__((ext_vector_type(4)));
typedef int          i32x4 __attribute__((ext_vector_type(4)));
typedef int          i32x8 __attribute__((ext_vector_type(8)));

#define B_DIM 2
#define T_DIM 2048
#define C_DIM 2048
#define NH    16
#define HS    128
#define DL    1024
#define MT    (B_DIM * T_DIM)   // 4096 rows

#define WMMA_BF16(a, b, c) \
  __builtin_amdgcn_wmma_f32_16x16x32_bf16(false, (a), false, (b), (short)0, (c), false, false)

#if defined(__gfx1250__) && __has_builtin(__builtin_amdgcn_tensor_load_to_lds) && \
    __has_builtin(__builtin_amdgcn_s_wait_tensorcnt)
#define USE_TDM 1
#else
#define USE_TDM 0
#endif

#if USE_TDM
// Flat addresses of LDS objects carry the LDS byte offset in the low 32 bits
// (ISA 10.2 aperture mapping).
__device__ __forceinline__ uint32_t lds_off(const void* p) {
  return (uint32_t)(uintptr_t)p;
}

// 2-D TDM tile load: tile_h rows of tile_w 4-byte units, row pitch
// row_pitch_units, from gsrc into contiguous LDS at lds_byte.
// D# layout per CDNA5 ISA (data_size=2 -> 4B).
__device__ __forceinline__ void tdm_load_2d(const void* gsrc, uint32_t lds_byte,
                                            uint32_t row_units, uint32_t nrows,
                                            uint32_t row_pitch_units,
                                            uint32_t tile_w_units, uint32_t tile_h) {
  uint64_t ga = (uint64_t)(uintptr_t)gsrc;
  u32x4 g0;
  g0[0] = 1u;                                        // count=1, user descriptor
  g0[1] = lds_byte;                                  // lds_addr
  g0[2] = (uint32_t)ga;                              // global_addr[31:0]
  g0[3] = (uint32_t)(ga >> 32) | (2u << 30);         // addr[56:32] | type=2
  i32x8 g1;
  g1[0] = (int)(2u << 16);                           // wg_mask=0, data_size=4B
  g1[1] = (int)((row_units & 0xFFFFu) << 16);        // tensor_dim0 lo16
  g1[2] = (int)((row_units >> 16) | ((nrows & 0xFFFFu) << 16));   // d0 hi | d1 lo
  g1[3] = (int)((nrows >> 16) | (tile_w_units << 16));            // d1 hi | tile_dim0
  g1[4] = (int)(tile_h & 0xFFFFu);                   // tile_dim1 | tile_dim2=0
  g1[5] = (int)row_pitch_units;                      // tensor_dim0_stride lo32
  g1[6] = 0;                                         // stride0 hi | stride1 lo
  g1[7] = 0;
  i32x4 z4 = {0, 0, 0, 0};
#if __clang_major__ >= 23
  i32x8 z8 = {0, 0, 0, 0, 0, 0, 0, 0};
  __builtin_amdgcn_tensor_load_to_lds(g0, g1, z4, z4, z8, 0);
#else
  __builtin_amdgcn_tensor_load_to_lds(g0, g1, z4, z4, 0);
#endif
}
#endif  // USE_TDM

// A-matrix fragment, 16x32 bf16 (ISA 7.12.2 "16-bit A-Matrix 16x32"):
// lane holds row M=lane&15; elems 0..7 -> K = half*8 + e, elems 8..15 -> K = 16 + half*8 + (e-8).
__device__ __forceinline__ v16bf load_fragA(const bf16_t* rowbase, int half) {
  bf16x8 lo = *(const bf16x8*)(rowbase + half * 8);
  bf16x8 hi = *(const bf16x8*)(rowbase + 16 + half * 8);
  v16bf r;
#pragma unroll
  for (int i = 0; i < 8; ++i) { r[i] = lo[i]; r[i + 8] = hi[i]; }
  return r;
}

// B-matrix fragment, 32x16 bf16 (dense analog of ISA 7.12.4 B layout):
// lane holds col N=lane&15; elems 0..7 -> K = half*8 + e, elems 8..15 -> K = 16 + half*8 + (e-8).
__device__ __forceinline__ v16bf load_fragB(const bf16_t* colbase, int half) {
  bf16x8 lo = *(const bf16x8*)(colbase + half * 8);
  bf16x8 hi = *(const bf16x8*)(colbase + 16 + half * 8);
  v16bf r;
#pragma unroll
  for (int i = 0; i < 8; ++i) { r[i] = lo[i]; r[i + 8] = hi[i]; }
  return r;
}

// ---------------------------------------------------------------------------
// f32 -> bf16 cast (vectorized x4)
// ---------------------------------------------------------------------------
__global__ __launch_bounds__(256) void cast_f32_bf16x4(const float4* __restrict__ in,
                                                       bf16x4* __restrict__ out, int n4) {
  int i = blockIdx.x * 256 + threadIdx.x;
  if (i < n4) {
    float4 v = in[i];
    bf16x4 o;
    o[0] = (bf16_t)v.x; o[1] = (bf16_t)v.y; o[2] = (bf16_t)v.z; o[3] = (bf16_t)v.w;
    out[i] = o;
  }
}

// ---------------------------------------------------------------------------
// bf16 GEMM: C[M,N] = A[M,K] @ W[K,N] (+bias). Block tile 128x128, BK=32,
// double-buffered LDS. 8 waves (wave32): 4x2, each wave 32x64 = 2x4 frags.
// A-tile staged by TDM (wave 0), B-tile register-staged + transposed.
// ---------------------------------------------------------------------------
template <bool OUT_BF16>
__global__ __launch_bounds__(256) void gemm_bf16(const bf16_t* __restrict__ A,
                                                 const bf16_t* __restrict__ W,
                                                 void* __restrict__ Cout,
                                                 const float* __restrict__ bias,
                                                 int M, int N, int K) {
  __shared__ bf16_t As[2][128 * 32];   // [buf][row][k]
  __shared__ bf16_t Bs[2][128 * 32];   // [buf][n][k] (transposed)

  const int tid  = threadIdx.x;
  const int lane = tid & 31, wid = tid >> 5;
  const int half = (lane >> 4) & 1, l16 = lane & 15;
  const int wm = wid >> 1, wn = wid & 1;
  const int bm = blockIdx.y * 128, bn = blockIdx.x * 128;

  v8f acc[2][4] = {};

  // ---- prologue: stage k0 = 0 into buffer 0 --------------------------------
#if USE_TDM
  if (wid == 0)
    tdm_load_2d(A + (size_t)bm * K, lds_off(&As[0][0]),
                (uint32_t)(K >> 1), (uint32_t)M, (uint32_t)(K >> 1), 16u, 128u);
#else
#pragma unroll
  for (int i = 0; i < 2; ++i) {
    int j = tid + i * 256;
    int r = j >> 2, c = (j & 3) * 8;
    *(bf16x8*)(&As[0][r * 32 + c]) = *(const bf16x8*)(A + (size_t)(bm + r) * K + c);
  }
#endif
#pragma unroll
  for (int i = 0; i < 2; ++i) {
    int j = tid + i * 256;
    int r = j >> 4, c = (j & 15) * 8;
    bf16x8 v = *(const bf16x8*)(W + (size_t)r * N + bn + c);
#pragma unroll
    for (int e = 0; e < 8; ++e) Bs[0][(c + e) * 32 + r] = v[e];
  }
#if USE_TDM
  if (wid == 0) __builtin_amdgcn_s_wait_tensorcnt(0);
#endif
  __syncthreads();

  int cur = 0;
  for (int k0 = 0; k0 < K; k0 += 32) {
    const int nxt = cur ^ 1;
    const bool have_next = (k0 + 32) < K;

    // ---- stage next tile: TDM for A, register-stage B ----------------------
    bf16x8 bstage[2];
#if !USE_TDM
    bf16x8 astage[2];
#endif
    if (have_next) {
#if USE_TDM
      if (wid == 0)
        tdm_load_2d(A + (size_t)bm * K + (k0 + 32), lds_off(&As[nxt][0]),
                    (uint32_t)(K >> 1), (uint32_t)M, (uint32_t)(K >> 1), 16u, 128u);
#else
#pragma unroll
      for (int i = 0; i < 2; ++i) {
        int j = tid + i * 256;
        int r = j >> 2, c = (j & 3) * 8;
        astage[i] = *(const bf16x8*)(A + (size_t)(bm + r) * K + (k0 + 32) + c);
      }
#endif
#pragma unroll
      for (int i = 0; i < 2; ++i) {
        int j = tid + i * 256;
        int r = j >> 4, c = (j & 15) * 8;
        bstage[i] = *(const bf16x8*)(W + (size_t)(k0 + 32 + r) * N + bn + c);
      }
    }

    // ---- compute on current buffers ---------------------------------------
    v16bf af[2], bfr[4];
#pragma unroll
    for (int i = 0; i < 2; ++i)
      af[i] = load_fragA(&As[cur][(wm * 32 + i * 16 + l16) * 32], half);
#pragma unroll
    for (int j = 0; j < 4; ++j)
      bfr[j] = load_fragB(&Bs[cur][(wn * 64 + j * 16 + l16) * 32], half);
#pragma unroll
    for (int i = 0; i < 2; ++i)
#pragma unroll
      for (int j = 0; j < 4; ++j)
        acc[i][j] = WMMA_BF16(af[i], bfr[j], acc[i][j]);

    // ---- commit staged tiles, sync, flip ----------------------------------
    if (have_next) {
#if !USE_TDM
#pragma unroll
      for (int i = 0; i < 2; ++i) {
        int j = tid + i * 256;
        int r = j >> 2, c = (j & 3) * 8;
        *(bf16x8*)(&As[nxt][r * 32 + c]) = astage[i];
      }
#endif
#pragma unroll
      for (int i = 0; i < 2; ++i) {
        int j = tid + i * 256;
        int r = j >> 4, c = (j & 15) * 8;
#pragma unroll
        for (int e = 0; e < 8; ++e) Bs[nxt][(c + e) * 32 + r] = bstage[i][e];
      }
    }
#if USE_TDM
    if (wid == 0 && have_next) __builtin_amdgcn_s_wait_tensorcnt(0);
#endif
    __syncthreads();
    cur = nxt;
  }

#pragma unroll
  for (int i = 0; i < 2; ++i)
#pragma unroll
    for (int j = 0; j < 4; ++j) {
      int col = bn + wn * 64 + j * 16 + l16;
      float bv = bias ? bias[col] : 0.0f;
#pragma unroll
      for (int r = 0; r < 8; ++r) {
        int row = bm + wm * 32 + i * 16 + r + 8 * half;
        float v = acc[i][j][r] + bv;
        if (OUT_BF16)
          ((bf16_t*)Cout)[(size_t)row * N + col] = (bf16_t)v;
        else
          ((float*)Cout)[(size_t)row * N + col] = v;
      }
    }
}

// ---------------------------------------------------------------------------
// Flash attention: block = (b, h, 128-query tile), 8 waves x 16 query rows.
// Double-buffered 64-key K/V tiles; K staged by TDM, V register-staged with
// a transposed LDS commit; online softmax; all matmuls WMMA.
// ---------------------------------------------------------------------------
__global__ __launch_bounds__(256) void mla_attention(const bf16_t* __restrict__ Q,
                                                     const bf16_t* __restrict__ Kg,
                                                     const bf16_t* __restrict__ Vg,
                                                     bf16_t* __restrict__ Y) {
  __shared__ bf16_t Ks[2][64 * 128];   // [buf][key][dim]   32 KB
  __shared__ bf16_t VT[2][128 * 64];   // [buf][dim][key]   32 KB
  __shared__ bf16_t Ps[8][16 * 64];    // per-wave P tiles  16 KB

  const int tid  = threadIdx.x;
  const int lane = tid & 31, wid = tid >> 5;
  const int half = (lane >> 4) & 1, l16 = lane & 15;

  const int blk = blockIdx.x;
  const int qt = blk & 15;          // T/128 = 16 query tiles
  const int h  = (blk >> 4) & 15;   // 16 heads
  const int b  = blk >> 8;          // batch
  const int qbase = qt * 128 + wid * 16;
  const size_t rowbase = (size_t)b * T_DIM;
  const size_t headoff = (size_t)h * HS;

  // Load this wave's Q fragments (16 rows x 128 dims = 4 A-frags), once.
  v16bf Qf[4];
  {
    const bf16_t* qp = Q + (rowbase + qbase + l16) * C_DIM + headoff;
#pragma unroll
    for (int kc = 0; kc < 4; ++kc) Qf[kc] = load_fragA(qp + kc * 32, half);
  }

  v8f O[8] = {};
  float mrow[8], lrow[8];
#pragma unroll
  for (int r = 0; r < 8; ++r) { mrow[r] = -1e30f; lrow[r] = 0.0f; }

  const float scale = 0.08838834764831845f;  // 1/sqrt(128)
  const int nkt = 2 * qt + 2;                // key tiles under the causal mask

  // ---- prologue: stage key tile 0 into buffer 0 ---------------------------
#if USE_TDM
  if (wid == 0)
    tdm_load_2d(Kg + rowbase * C_DIM + headoff, lds_off(&Ks[0][0]),
                (uint32_t)(C_DIM >> 1), (uint32_t)MT, (uint32_t)(C_DIM >> 1), 64u, 64u);
#else
#pragma unroll
  for (int i = 0; i < 4; ++i) {
    int j = tid + i * 256;
    int row = j >> 4, cc = (j & 15) * 8;
    *(bf16x8*)(&Ks[0][row * 128 + cc]) =
        *(const bf16x8*)(Kg + (rowbase + row) * C_DIM + headoff + cc);
  }
#endif
#pragma unroll
  for (int i = 0; i < 4; ++i) {
    int j = tid + i * 256;
    int row = j >> 4, cc = (j & 15) * 8;
    bf16x8 vv = *(const bf16x8*)(Vg + (rowbase + row) * C_DIM + headoff + cc);
#pragma unroll
    for (int e = 0; e < 8; ++e) VT[0][(cc + e) * 64 + row] = vv[e];
  }
#if USE_TDM
  if (wid == 0) __builtin_amdgcn_s_wait_tensorcnt(0);
#endif
  __syncthreads();

  int cur = 0;
  for (int kt = 0; kt < nkt; ++kt) {
    const int kb = kt * 64;
    const int nxt = cur ^ 1;
    const bool have_next = (kt + 1) < nkt;

    // ---- stage next key tile ----------------------------------------------
    bf16x8 vstage[4];
#if !USE_TDM
    bf16x8 kstage[4];
#endif
    if (have_next) {
      const int kb2 = kb + 64;
#if USE_TDM
      if (wid == 0)
        tdm_load_2d(Kg + (rowbase + kb2) * C_DIM + headoff, lds_off(&Ks[nxt][0]),
                    (uint32_t)(C_DIM >> 1), (uint32_t)MT, (uint32_t)(C_DIM >> 1), 64u, 64u);
#else
#pragma unroll
      for (int i = 0; i < 4; ++i) {
        int j = tid + i * 256;
        int row = j >> 4, cc = (j & 15) * 8;
        kstage[i] = *(const bf16x8*)(Kg + (rowbase + kb2 + row) * C_DIM + headoff + cc);
      }
#endif
#pragma unroll
      for (int i = 0; i < 4; ++i) {
        int j = tid + i * 256;
        int row = j >> 4, cc = (j & 15) * 8;
        vstage[i] = *(const bf16x8*)(Vg + (rowbase + kb2 + row) * C_DIM + headoff + cc);
      }
    }

    // ---- S = Q @ K^T  (16x64) ---------------------------------------------
    v8f S[4] = {};
#pragma unroll
    for (int kc = 0; kc < 4; ++kc) {
#pragma unroll
      for (int nf = 0; nf < 4; ++nf) {
        v16bf kf = load_fragB(&Ks[cur][(nf * 16 + l16) * 128 + kc * 32], half);
        S[nf] = WMMA_BF16(Qf[kc], kf, S[nf]);
      }
    }

    // ---- scale, causal mask, row max (rows live in a half-wave) -----------
    float rmax[8];
#pragma unroll
    for (int r = 0; r < 8; ++r) rmax[r] = -1e30f;
#pragma unroll
    for (int nf = 0; nf < 4; ++nf) {
      int kidx = kb + nf * 16 + l16;
#pragma unroll
      for (int r = 0; r < 8; ++r) {
        int qidx = qbase + r + 8 * half;
        float s = S[nf][r] * scale;
        if (kidx > qidx) s = -1e30f;
        S[nf][r] = s;
        rmax[r] = fmaxf(rmax[r], s);
      }
    }
#pragma unroll
    for (int r = 0; r < 8; ++r) {
      float v = rmax[r];
      v = fmaxf(v, __shfl_xor(v, 1, 32));
      v = fmaxf(v, __shfl_xor(v, 2, 32));
      v = fmaxf(v, __shfl_xor(v, 4, 32));
      v = fmaxf(v, __shfl_xor(v, 8, 32));
      rmax[r] = v;
    }

    float alpha[8];
#pragma unroll
    for (int r = 0; r < 8; ++r) {
      float mnew = fmaxf(mrow[r], rmax[r]);
      alpha[r] = __expf(mrow[r] - mnew);
      mrow[r] = mnew;
    }

    // ---- P = exp(S - m), row sums, stage P in per-wave LDS ----------------
    float rsum[8];
#pragma unroll
    for (int r = 0; r < 8; ++r) rsum[r] = 0.0f;
#pragma unroll
    for (int nf = 0; nf < 4; ++nf)
#pragma unroll
      for (int r = 0; r < 8; ++r) {
        float p = __expf(S[nf][r] - mrow[r]);
        rsum[r] += p;
        Ps[wid][(r + 8 * half) * 64 + nf * 16 + l16] = (bf16_t)p;
      }
#pragma unroll
    for (int r = 0; r < 8; ++r) {
      float v = rsum[r];
      v += __shfl_xor(v, 1, 32);
      v += __shfl_xor(v, 2, 32);
      v += __shfl_xor(v, 4, 32);
      v += __shfl_xor(v, 8, 32);
      lrow[r] = lrow[r] * alpha[r] + v;
    }

    // ---- rescale running O, then O += P @ V -------------------------------
#pragma unroll
    for (int of = 0; of < 8; ++of)
#pragma unroll
      for (int r = 0; r < 8; ++r) O[of][r] *= alpha[r];

#pragma unroll
    for (int kc2 = 0; kc2 < 2; ++kc2) {
      v16bf pf = load_fragA(&Ps[wid][l16 * 64 + kc2 * 32], half);
#pragma unroll
      for (int of = 0; of < 8; ++of) {
        v16bf vf = load_fragB(&VT[cur][(of * 16 + l16) * 64 + kc2 * 32], half);
        O[of] = WMMA_BF16(pf, vf, O[of]);
      }
    }

    // ---- commit staged V (+K fallback), sync, flip ------------------------
    if (have_next) {
#if !USE_TDM
#pragma unroll
      for (int i = 0; i < 4; ++i) {
        int j = tid + i * 256;
        int row = j >> 4, cc = (j & 15) * 8;
        *(bf16x8*)(&Ks[nxt][row * 128 + cc]) = kstage[i];
      }
#endif
#pragma unroll
      for (int i = 0; i < 4; ++i) {
        int j = tid + i * 256;
        int row = j >> 4, cc = (j & 15) * 8;
#pragma unroll
        for (int e = 0; e < 8; ++e) VT[nxt][(cc + e) * 64 + row] = vstage[i][e];
      }
    }
#if USE_TDM
    if (wid == 0 && have_next) __builtin_amdgcn_s_wait_tensorcnt(0);
#endif
    __syncthreads();
    cur = nxt;
  }

  // ---- normalize and write y (bf16), heads interleaved in (B,T,C) ---------
#pragma unroll
  for (int of = 0; of < 8; ++of)
#pragma unroll
    for (int r = 0; r < 8; ++r) {
      int qrow = qbase + r + 8 * half;
      Y[(rowbase + qrow) * C_DIM + headoff + of * 16 + l16] =
          (bf16_t)(O[of][r] / lrow[r]);
    }
}

// ---------------------------------------------------------------------------
// Launch
// ---------------------------------------------------------------------------
extern "C" void kernel_launch(void* const* d_in, const int* in_sizes, int n_in,
                              void* d_out, int out_size, void* d_ws, size_t ws_size,
                              hipStream_t stream) {
  (void)in_sizes; (void)n_in; (void)out_size; (void)ws_size;

  const float* x  = (const float*)d_in[0];
  const float* Wq = (const float*)d_in[1];
  const float* Wc = (const float*)d_in[2];
  const float* Wk = (const float*)d_in[3];
  const float* Wv = (const float*)d_in[4];
  const float* Wo = (const float*)d_in[5];
  const float* bo = (const float*)d_in[6];
  float* out = (float*)d_out;

  char* ws = (char*)d_ws;
  auto carve = [&](size_t elems) {
    bf16_t* p = (bf16_t*)ws;
    ws += (elems * sizeof(bf16_t) + 255) & ~(size_t)255;
    return p;
  };

  const size_t nX  = (size_t)MT * C_DIM;
  const size_t nWq = (size_t)C_DIM * C_DIM;
  const size_t nWc = (size_t)C_DIM * DL;
  const size_t nWk = (size_t)DL * C_DIM;
  const size_t nCL = (size_t)MT * DL;

  bf16_t* xb  = carve(nX);
  bf16_t* Wqb = carve(nWq);
  bf16_t* Wcb = carve(nWc);
  bf16_t* Wkb = carve(nWk);
  bf16_t* Wvb = carve(nWk);
  bf16_t* Wob = carve(nWq);
  bf16_t* qb  = carve(nX);
  bf16_t* clb = carve(nCL);
  bf16_t* kb  = carve(nX);
  bf16_t* vb  = carve(nX);
  bf16_t* yb  = carve(nX);

  auto cast4 = [&](const float* src, bf16_t* dst, size_t n) {
    int n4 = (int)(n / 4);
    cast_f32_bf16x4<<<(n4 + 255) / 256, 256, 0, stream>>>((const float4*)src,
                                                          (bf16x4*)dst, n4);
  };
  cast4(x,  xb,  nX);
  cast4(Wq, Wqb, nWq);
  cast4(Wc, Wcb, nWc);
  cast4(Wk, Wkb, nWk);
  cast4(Wv, Wvb, nWk);
  cast4(Wo, Wob, nWq);

  // q = x @ Wq ; c_latent = x @ Wc ; k = c @ Wk ; v = c @ Wv   (bf16 outputs)
  gemm_bf16<true><<<dim3(C_DIM / 128, MT / 128), 256, 0, stream>>>(
      xb, Wqb, qb, nullptr, MT, C_DIM, C_DIM);
  gemm_bf16<true><<<dim3(DL / 128, MT / 128), 256, 0, stream>>>(
      xb, Wcb, clb, nullptr, MT, DL, C_DIM);
  gemm_bf16<true><<<dim3(C_DIM / 128, MT / 128), 256, 0, stream>>>(
      clb, Wkb, kb, nullptr, MT, C_DIM, DL);
  gemm_bf16<true><<<dim3(C_DIM / 128, MT / 128), 256, 0, stream>>>(
      clb, Wvb, vb, nullptr, MT, C_DIM, DL);

  // Flash attention: B * H * (T/128) = 512 blocks.
  mla_attention<<<B_DIM * NH * (T_DIM / 128), 256, 0, stream>>>(qb, kb, vb, yb);

  // out = y @ Wo + bo  (f32 output)
  gemm_bf16<false><<<dim3(C_DIM / 128, MT / 128), 256, 0, stream>>>(
      yb, Wob, (void*)out, bo, MT, C_DIM, C_DIM);
}
